// LogitSignalTransformer_17162689315438
// MI455X (gfx1250) — matrix operand order repaired and checked by reading, not compiled, not run on hardware
//
#include <hip/hip_runtime.h>
#include <hip/hip_bf16.h>
#include <math.h>

// ---------------------------------------------------------------------------
// LogitSignalTransformer for MI455X (gfx1250, wave32, WMMA + async-LDS)
//
//   pack all weights -> WMMA B-fragment layout (bf16, 32B/lane contiguous)
//   features kernel  -> rowmax, Z_T, entropies, gumbel samples, cons, pos
//   G1: logits(f32->bf16 on the fly) @ w1   [2048x32000x1024]  WMMA bf16
//       (B panel staged in LDS via GLOBAL_LOAD_ASYNC_TO_LDS_B128, double buf)
//   LN+GELU -> h1b(bf16)
//   G2: h1b @ w2 [2048x1024x256] -> LN+GELU -> combined[:,0:256] (bf16)
//   G3: combined(2048x320pad) @ fus_w -> LN+GELU -> X(f32), Xb(bf16)
//   3 transformer layers with seqlen==1 (attention == v-projection)
//   heads -> d_out = [hall(2048) | sigmoid(2048) | scale(2048x5)]
// ---------------------------------------------------------------------------

typedef __attribute__((ext_vector_type(16))) __bf16 bf16x16;
typedef __attribute__((ext_vector_type(8)))  __bf16 bf16x8;
typedef __attribute__((ext_vector_type(8)))  float  f32x8;
typedef __attribute__((ext_vector_type(4)))  float  f32x4;
typedef __attribute__((ext_vector_type(4)))  int    i32x4;

constexpr int B_    = 2048;
constexpr int V_    = 32000;
constexpr int CD_   = 256;
constexpr int HID_  = 128;
constexpr int NL_   = 3;
constexpr int NS_   = 10;
constexpr int MAXLEN_ = 512;
constexpr int FEAT_ = 284;   // CD + 5 + 4 + 3 + 16
constexpr int FPAD_ = 320;   // padded so every GEMM K is a multiple of 64

static __device__ __forceinline__ float gelu_f(float x) {
    return 0.5f * x * (1.0f + erff(x * 0.70710678118654752440f));
}
static __device__ __forceinline__ unsigned hashu(unsigned x) {
    x ^= x >> 16; x *= 0x7feb352dU;
    x ^= x >> 15; x *= 0x846ca68bU;
    x ^= x >> 16; return x;
}

// ---- CDNA5 async global->LDS copy (16B per lane), with sync fallback ------
#if __has_builtin(__builtin_amdgcn_global_load_async_to_lds_b128)
#define HAVE_ASYNC_LDS 1
static __device__ __forceinline__ void async_cp16(const void* g, void* l) {
    __builtin_amdgcn_global_load_async_to_lds_b128(
        (__attribute__((address_space(1))) i32x4*)(g),
        (__attribute__((address_space(3))) i32x4*)(l), 0, 0);
}
static __device__ __forceinline__ void wait_async_le2() {
#if __has_builtin(__builtin_amdgcn_s_wait_asynccnt)
    __builtin_amdgcn_s_wait_asynccnt(2);
#else
    asm volatile("s_wait_asynccnt 0x2" ::: "memory");
#endif
}
static __device__ __forceinline__ void wait_async_le0() {
#if __has_builtin(__builtin_amdgcn_s_wait_asynccnt)
    __builtin_amdgcn_s_wait_asynccnt(0);
#else
    asm volatile("s_wait_asynccnt 0x0" ::: "memory");
#endif
}
#else
#define HAVE_ASYNC_LDS 0
static __device__ __forceinline__ void async_cp16(const void* g, void* l) {
    *(uint4*)l = *(const uint4*)g;   // sync fallback: vmem load + ds store
}
static __device__ __forceinline__ void wait_async_le2() {}
static __device__ __forceinline__ void wait_async_le0() {}
#endif

// ---------------------------------------------------------------------------
// Pack f32 weight matrix (row-major KxN, row stride ldw, column window col0)
// into WMMA B-fragment order: frag = (ktile * ntiles + ntile), per lane 8
// contiguous dwords (16 bf16 = K 16*(lane/16)+0..15 of its column).  Rows k >= Ksrc pad 0.
// ---------------------------------------------------------------------------
__global__ void pack_b_kernel(const float* __restrict__ W, int ldw, int col0,
                              int Kpad, int Ksrc, int N,
                              unsigned* __restrict__ out) {
    int t = blockIdx.x * blockDim.x + threadIdx.x;
    int total = (Kpad >> 5) * (N >> 4) * 256;
    if (t >= total) return;
    int frag = t >> 8;
    int r    = t & 255;
    int lane = r >> 3;
    int dw   = r & 7;
    int ntiles = N >> 4;
    int ktile = frag / ntiles;
    int ntile = frag - ktile * ntiles;
    int n = (ntile << 4) + (lane & 15);
    int k = (ktile << 5) + ((lane & 16) ? 16 : 0) + (dw << 1);
    float lo = (k     < Ksrc) ? W[(size_t)k       * ldw + col0 + n] : 0.0f;
    float hi = (k + 1 < Ksrc) ? W[(size_t)(k + 1) * ldw + col0 + n] : 0.0f;
    __bf16 blo = (__bf16)lo;
    __bf16 bhi = (__bf16)hi;
    unsigned ulo = __builtin_bit_cast(unsigned short, blo);
    unsigned uhi = __builtin_bit_cast(unsigned short, bhi);
    out[t] = (uhi << 16) | ulo;
}

// ---------------------------------------------------------------------------
// WMMA GEMM: C[MxN] = A[MxK] * Bpacked + (bias?).  A row-major (f32 or bf16).
// Block = 256 threads = 8 waves (4 along M, 2 along N), tile 128x64.
// B panel (64-deep K slab = 2 WMMA k-steps, 4 ntiles, 8KB) is staged in LDS
// via async copies, double buffered; all 4 M-waves share it.
// Requires: M%128==0, N%64==0, K%64==0 (true at every call site).
// ---------------------------------------------------------------------------
template <bool A_IS_F32>
__global__ void wmma_gemm_kernel(const void* __restrict__ Av, int lda,
                                 const unsigned* __restrict__ Bp,
                                 float* __restrict__ C,
                                 const float* __restrict__ bias,
                                 int M, int N, int K) {
    __shared__ __align__(32) unsigned sB[2][2048];   // 2 x 8KB

    const int tid  = threadIdx.x;
    const int lane = tid & 31;
    const int wave = tid >> 5;
    const int wm = wave & 3;
    const int wn = wave >> 2;
    const int m_base = blockIdx.y * 128 + wm * 32;
    const int n_base = blockIdx.x * 64 + wn * 32;

    const int ntiles = N >> 4;
    const int nblk   = (blockIdx.x * 64) >> 4;   // first ntile of this block

    // per-thread slice of the 8KB slab: chunk = k-step (0/1), 8 dwords each
    const int chunk = tid >> 7;              // 0 or 1
    const int coff  = (tid & 127) << 3;      // dword offset within 4KB chunk

    f32x8 acc00 = {}, acc01 = {}, acc10 = {}, acc11 = {};
    const int mrow0  = m_base + (lane & 15);
    const int mrow1  = mrow0 + 16;
    const int kb_off = (lane & 16) ? 16 : 0;

    auto issue = [&](int buf, int k0) {
        const unsigned* g =
            Bp + ((size_t)((k0 >> 5) + chunk) * ntiles + nblk) * 256 + coff;
        unsigned* l = &sB[buf][chunk * 1024 + coff];
        async_cp16(g, l);
        async_cp16(g + 4, l + 4);
    };

    issue(0, 0);
    int cur = 0;
    for (int k0 = 0; k0 < K; k0 += 64) {
        const bool more = (k0 + 64) < K;
        if (more) { issue(cur ^ 1, k0 + 64); wait_async_le2(); }
        else      { wait_async_le0(); }
        __syncthreads();

#pragma unroll
        for (int j = 0; j < 2; j++) {
            const int kk = k0 + j * 32;
            bf16x16 a0, a1;
            if (A_IS_F32) {
                const float* A = (const float*)Av;
                const float* p0 = A + (size_t)mrow0 * lda + kk + kb_off;
                const float* p1 = A + (size_t)mrow1 * lda + kk + kb_off;
                f32x4 c00 = *(const f32x4*)(p0);
                f32x4 c01 = *(const f32x4*)(p0 + 4);
                f32x4 c02 = *(const f32x4*)(p0 + 8);
                f32x4 c03 = *(const f32x4*)(p0 + 12);
                f32x4 c10 = *(const f32x4*)(p1);
                f32x4 c11 = *(const f32x4*)(p1 + 4);
                f32x4 c12 = *(const f32x4*)(p1 + 8);
                f32x4 c13 = *(const f32x4*)(p1 + 12);
#pragma unroll
                for (int i = 0; i < 4; i++) {
                    a0[i]      = (__bf16)c00[i];
                    a0[4 + i]  = (__bf16)c01[i];
                    a0[8 + i]  = (__bf16)c02[i];
                    a0[12 + i] = (__bf16)c03[i];
                    a1[i]      = (__bf16)c10[i];
                    a1[4 + i]  = (__bf16)c11[i];
                    a1[8 + i]  = (__bf16)c12[i];
                    a1[12 + i] = (__bf16)c13[i];
                }
            } else {
                const __bf16* A = (const __bf16*)Av;
                const __bf16* p0 = A + (size_t)mrow0 * lda + kk + kb_off;
                const __bf16* p1 = A + (size_t)mrow1 * lda + kk + kb_off;
                bf16x8 l0 = *(const bf16x8*)(p0);
                bf16x8 h0 = *(const bf16x8*)(p0 + 8);
                bf16x8 l1 = *(const bf16x8*)(p1);
                bf16x8 h1 = *(const bf16x8*)(p1 + 8);
#pragma unroll
                for (int i = 0; i < 8; i++) {
                    a0[i] = l0[i]; a0[8 + i] = h0[i];
                    a1[i] = l1[i]; a1[8 + i] = h1[i];
                }
            }

            const unsigned* bb = &sB[cur][j * 1024];
            bf16x16 b0 = *(const bf16x16*)(bb + (wn * 2 + 0) * 256 + lane * 8);
            bf16x16 b1 = *(const bf16x16*)(bb + (wn * 2 + 1) * 256 + lane * 8);

            acc00 = __builtin_amdgcn_wmma_f32_16x16x32_bf16(
                false, a0, false, b0, (short)0, acc00, false, false);
            acc01 = __builtin_amdgcn_wmma_f32_16x16x32_bf16(
                false, a0, false, b1, (short)0, acc01, false, false);
            acc10 = __builtin_amdgcn_wmma_f32_16x16x32_bf16(
                false, a1, false, b0, (short)0, acc10, false, false);
            acc11 = __builtin_amdgcn_wmma_f32_16x16x32_bf16(
                false, a1, false, b1, (short)0, acc11, false, false);
        }
        __syncthreads();
        cur ^= 1;
    }

    const int n0 = n_base + (lane & 15);
    const int n1 = n0 + 16;
    const int ms = (lane & 16) ? 8 : 0;
    const float bi0 = bias ? bias[n0] : 0.0f;
    const float bi1 = bias ? bias[n1] : 0.0f;
#pragma unroll
    for (int r = 0; r < 8; r++) {
        C[(size_t)(m_base + ms + r)      * N + n0] = acc00[r] + bi0;
        C[(size_t)(m_base + ms + r)      * N + n1] = acc01[r] + bi1;
        C[(size_t)(m_base + 16 + ms + r) * N + n0] = acc10[r] + bi0;
        C[(size_t)(m_base + 16 + ms + r) * N + n1] = acc11[r] + bi1;
    }
}

// ---------------------------------------------------------------------------
// Per-row (bias + LayerNorm + GELU) -> bf16 (and optional f32) output.
// ---------------------------------------------------------------------------
__global__ void ln_gelu_kernel(const float* __restrict__ Xin, int ldx,
                               const float* __restrict__ bias,
                               const float* __restrict__ gam,
                               const float* __restrict__ bet,
                               int Ncols,
                               __bf16* __restrict__ outB, int ldob, int coloff,
                               float* __restrict__ outF, int ldof) {
    __shared__ float red[256];
    const int row = blockIdx.x;
    const int tid = threadIdx.x;
    const float* xr = Xin + (size_t)row * ldx;

    float s = 0.0f;
    for (int c = tid; c < Ncols; c += 256) s += xr[c] + bias[c];
    red[tid] = s; __syncthreads();
    for (int st = 128; st > 0; st >>= 1) {
        if (tid < st) red[tid] += red[tid + st];
        __syncthreads();
    }
    const float mean = red[0] / (float)Ncols;
    __syncthreads();

    float vs = 0.0f;
    for (int c = tid; c < Ncols; c += 256) {
        float d = xr[c] + bias[c] - mean;
        vs += d * d;
    }
    red[tid] = vs; __syncthreads();
    for (int st = 128; st > 0; st >>= 1) {
        if (tid < st) red[tid] += red[tid + st];
        __syncthreads();
    }
    const float rstd = rsqrtf(red[0] / (float)Ncols + 1e-5f);

    for (int c = tid; c < Ncols; c += 256) {
        float y = (xr[c] + bias[c] - mean) * rstd * gam[c] + bet[c];
        float z = gelu_f(y);
        outB[(size_t)row * ldob + coloff + c] = (__bf16)z;
        if (outF) outF[(size_t)row * ldof + c] = z;
    }
}

// ---------------------------------------------------------------------------
// Residual + LayerNorm over HID=128 cols: X = LN(X + T + bias); bf16 copy too.
// ---------------------------------------------------------------------------
__global__ void resid_ln_kernel(float* __restrict__ X,
                                const float* __restrict__ T,
                                const float* __restrict__ bias,
                                const float* __restrict__ gam,
                                const float* __restrict__ bet,
                                __bf16* __restrict__ Xb) {
    __shared__ float red[128];
    const int row = blockIdx.x;
    const int c = threadIdx.x;
    const size_t idx = (size_t)row * HID_ + c;
    float x = X[idx] + T[idx] + bias[c];
    red[c] = x; __syncthreads();
    for (int st = 64; st > 0; st >>= 1) {
        if (c < st) red[c] += red[c + st];
        __syncthreads();
    }
    const float mean = red[0] * (1.0f / HID_);
    __syncthreads();
    const float d = x - mean;
    red[c] = d * d; __syncthreads();
    for (int st = 64; st > 0; st >>= 1) {
        if (c < st) red[c] += red[c + st];
        __syncthreads();
    }
    const float rstd = rsqrtf(red[0] * (1.0f / HID_) + 1e-5f);
    const float y = d * rstd * gam[c] + bet[c];
    X[idx] = y;
    Xb[idx] = (__bf16)y;
}

// elementwise f32 -> (gelu?) -> bf16
__global__ void cast_act_kernel(const float* __restrict__ in, int total, int act,
                                __bf16* __restrict__ out) {
    int i = blockIdx.x * 256 + threadIdx.x;
    if (i >= total) return;
    float v = in[i];
    if (act) v = gelu_f(v);
    out[i] = (__bf16)v;
}

// ---------------------------------------------------------------------------
// Per-row entropy / sampling / consistency features.  One 256-thread WG/row,
// 3 streaming passes (max | Z_T | entropy+gumbel-argmax).
// Writes combined[:, 256:320] (entropy5, grads4, cons3, pos16, zero pad).
// ---------------------------------------------------------------------------
__global__ void features_kernel(const float* __restrict__ logits,
                                const int* __restrict__ positions,
                                const float* __restrict__ pos_emb,
                                __bf16* __restrict__ comb) {
    __shared__ float redf[256];
    __shared__ int   redi[256];
    __shared__ float Zsh[5];
    __shared__ float entsh[5];
    __shared__ int   sampsh[NS_];

    const int row = blockIdx.x;
    const int tid = threadIdx.x;
    const float* lr = logits + (size_t)row * V_;
    const float invT[5] = {1.0f / 0.3f, 2.0f, 1.0f, 1.0f / 1.5f, 0.5f};

    // zero the padded feature columns [FEAT_, FPAD_)
    for (int c = FEAT_ + tid; c < FPAD_; c += 256)
        comb[(size_t)row * FPAD_ + c] = (__bf16)0.0f;

    // pass 1: max
    float mx = -3.4e38f;
    for (int i = tid; i < V_; i += 256) mx = fmaxf(mx, lr[i]);
    redf[tid] = mx; __syncthreads();
    for (int st = 128; st > 0; st >>= 1) {
        if (tid < st) redf[tid] = fmaxf(redf[tid], redf[tid + st]);
        __syncthreads();
    }
    const float rmax = redf[0];
    __syncthreads();

    // pass 2: partition functions per temperature
    float zs[5] = {0, 0, 0, 0, 0};
    for (int i = tid; i < V_; i += 256) {
        const float d = lr[i] - rmax;
#pragma unroll
        for (int t = 0; t < 5; t++) zs[t] += __expf(d * invT[t]);
    }
#pragma unroll
    for (int t = 0; t < 5; t++) {
        redf[tid] = zs[t]; __syncthreads();
        for (int st = 128; st > 0; st >>= 1) {
            if (tid < st) redf[tid] += redf[tid + st];
            __syncthreads();
        }
        if (tid == 0) Zsh[t] = redf[0];
        __syncthreads();
    }

    // pass 3: clamped entropy + 10 gumbel-argmax samples
    float ent[5] = {0, 0, 0, 0, 0};
    float bestv[NS_];
    int   besti[NS_];
    unsigned hseed[NS_];
#pragma unroll
    for (int s = 0; s < NS_; s++) {
        bestv[s] = -3.4e38f;
        besti[s] = 0;
        hseed[s] = hashu(0x2a2a2a2aU ^ (unsigned)(row * 131 + s * 7919));
    }
    const float zinv[5] = {1.0f / Zsh[0], 1.0f / Zsh[1], 1.0f / Zsh[2],
                           1.0f / Zsh[3], 1.0f / Zsh[4]};
    for (int i = tid; i < V_; i += 256) {
        const float l = lr[i];
        const float d = l - rmax;
#pragma unroll
        for (int t = 0; t < 5; t++) {
            float p = __expf(d * invT[t]) * zinv[t];
            p = fmaxf(p, 1e-10f);
            ent[t] -= p * __logf(p);
        }
        const unsigned hi = (unsigned)i * 0x9E3779B9U;
#pragma unroll
        for (int s = 0; s < NS_; s++) {
            unsigned h = hashu(hseed[s] ^ hi);
            float u = ((h >> 8) + 0.5f) * (1.0f / 16777216.0f);
            float g = -__logf(-__logf(u));
            float cand = l + g;
            if (cand > bestv[s]) { bestv[s] = cand; besti[s] = i; }
        }
    }
#pragma unroll
    for (int t = 0; t < 5; t++) {
        redf[tid] = ent[t]; __syncthreads();
        for (int st = 128; st > 0; st >>= 1) {
            if (tid < st) redf[tid] += redf[tid + st];
            __syncthreads();
        }
        if (tid == 0) entsh[t] = redf[0];
        __syncthreads();
    }
#pragma unroll
    for (int s = 0; s < NS_; s++) {
        redf[tid] = bestv[s]; redi[tid] = besti[s]; __syncthreads();
        for (int st = 128; st > 0; st >>= 1) {
            if (tid < st && redf[tid + st] > redf[tid]) {
                redf[tid] = redf[tid + st];
                redi[tid] = redi[tid + st];
            }
            __syncthreads();
        }
        if (tid == 0) sampsh[s] = redi[0];
        __syncthreads();
    }

    if (tid == 0) {
        float agree = 0.0f;
        for (int s = 0; s < NS_; s++) agree += (sampsh[s] == sampsh[0]) ? 1.0f : 0.0f;
        agree *= (1.0f / NS_);
        int uniq = 0;
        for (int i = 0; i < NS_; i++) {
            bool dup = false;
            for (int j = 0; j < i; j++) dup |= (sampsh[j] == sampsh[i]);
            if (!dup) uniq++;
        }
        const float unique_ratio = (float)uniq * (1.0f / NS_);
        const float top_prob = 1.0f / Zsh[2];   // T=1.0 slot

        float feats[28];
        for (int t = 0; t < 5; t++) feats[t] = entsh[t];
        for (int t = 0; t < 4; t++) feats[5 + t] = entsh[t + 1] - entsh[t];
        feats[9]  = agree;
        feats[10] = unique_ratio;
        feats[11] = agree - top_prob;
        int p = positions[row];
        p = p < 0 ? 0 : (p > MAXLEN_ - 1 ? MAXLEN_ - 1 : p);
        for (int j = 0; j < 16; j++) feats[12 + j] = pos_emb[p * 16 + j];

        __bf16* dst = comb + (size_t)row * FPAD_ + CD_;
        for (int j = 0; j < 28; j++) dst[j] = (__bf16)feats[j];
    }
}

// ---------------------------------------------------------------------------
// Heads: hall = gelu(x@W1+b1)@W2+b2 ; scale = gelu(x@Sw1+sb1)@Sw2+sb2
// d_out = [hall(2048) | sigmoid(2048) | scale(2048*5)]
// ---------------------------------------------------------------------------
__global__ void heads_kernel(const float* __restrict__ X,
                             const float* __restrict__ hw1, const float* __restrict__ hb1,
                             const float* __restrict__ hw2, const float* __restrict__ hb2,
                             const float* __restrict__ sw1, const float* __restrict__ sb1,
                             const float* __restrict__ sw2, const float* __restrict__ sb2,
                             float* __restrict__ out) {
    const int row = blockIdx.x * blockDim.x + threadIdx.x;
    if (row >= B_) return;
    const float* x = X + (size_t)row * HID_;

    float hall = hb2[0];
    for (int j = 0; j < 64; j++) {
        float a = hb1[j];
        for (int c = 0; c < HID_; c++) a += x[c] * hw1[c * 64 + j];
        hall += gelu_f(a) * hw2[j];
    }
    out[row] = hall;
    out[B_ + row] = 1.0f / (1.0f + __expf(-hall));

    float s32v[32];
    for (int j = 0; j < 32; j++) {
        float a = sb1[j];
        for (int c = 0; c < HID_; c++) a += x[c] * sw1[c * 32 + j];
        s32v[j] = gelu_f(a);
    }
    for (int t = 0; t < 5; t++) {
        float a = sb2[t];
        for (int j = 0; j < 32; j++) a += s32v[j] * sw2[j * 5 + t];
        out[2 * B_ + row * 5 + t] = a;
    }
}

// ---------------------------------------------------------------------------
extern "C" void kernel_launch(void* const* d_in, const int* in_sizes, int n_in,
                              void* d_out, int out_size, void* d_ws, size_t ws_size,
                              hipStream_t stream) {
    (void)in_sizes; (void)n_in; (void)out_size; (void)ws_size;

    const float* logits     = (const float*)d_in[0];
    const int*   positions  = (const int*)d_in[1];
    const float* comp_w1    = (const float*)d_in[2];
    const float* comp_b1    = (const float*)d_in[3];
    const float* comp_ln1_g = (const float*)d_in[4];
    const float* comp_ln1_b = (const float*)d_in[5];
    const float* comp_w2    = (const float*)d_in[6];
    const float* comp_b2    = (const float*)d_in[7];
    const float* comp_ln2_g = (const float*)d_in[8];
    const float* comp_ln2_b = (const float*)d_in[9];
    const float* pos_emb    = (const float*)d_in[10];
    const float* fus_w      = (const float*)d_in[11];
    const float* fus_b      = (const float*)d_in[12];
    const float* fus_ln_g   = (const float*)d_in[13];
    const float* fus_ln_b   = (const float*)d_in[14];
    const float* hall_w1 = (const float*)d_in[51];
    const float* hall_b1 = (const float*)d_in[52];
    const float* hall_w2 = (const float*)d_in[53];
    const float* hall_b2 = (const float*)d_in[54];
    const float* scale_w1 = (const float*)d_in[55];
    const float* scale_b1 = (const float*)d_in[56];
    const float* scale_w2 = (const float*)d_in[57];
    const float* scale_b2 = (const float*)d_in[58];

    float* out = (float*)d_out;

    // ---- workspace partition (256B aligned) ----
    char* wp = (char*)d_ws;
    auto alloc = [&](size_t bytes) -> void* {
        void* r = (void*)wp;
        wp += (bytes + 255) & ~(size_t)255;
        return r;
    };
    unsigned* bp_w1  = (unsigned*)alloc((size_t)(V_ / 32) * (1024 / 16) * 256 * 4);
    unsigned* bp_w2  = (unsigned*)alloc((size_t)(1024 / 32) * (CD_ / 16) * 256 * 4);
    unsigned* bp_fus = (unsigned*)alloc((size_t)(FPAD_ / 32) * (HID_ / 16) * 256 * 4);
    unsigned* bp_wv[NL_], *bp_wo[NL_], *bp_f1[NL_], *bp_f2[NL_];
    for (int l = 0; l < NL_; l++) {
        bp_wv[l] = (unsigned*)alloc((size_t)(HID_ / 32) * (HID_ / 16) * 256 * 4);
        bp_wo[l] = (unsigned*)alloc((size_t)(HID_ / 32) * (HID_ / 16) * 256 * 4);
        bp_f1[l] = (unsigned*)alloc((size_t)(HID_ / 32) * (512 / 16) * 256 * 4);
        bp_f2[l] = (unsigned*)alloc((size_t)(512 / 32) * (HID_ / 16) * 256 * 4);
    }
    float*  H1   = (float*)alloc((size_t)B_ * 1024 * 4);
    __bf16* h1b  = (__bf16*)alloc((size_t)B_ * 1024 * 2);
    __bf16* comb = (__bf16*)alloc((size_t)B_ * FPAD_ * 2);
    float*  X    = (float*)alloc((size_t)B_ * HID_ * 4);
    __bf16* Xb   = (__bf16*)alloc((size_t)B_ * HID_ * 2);
    float*  T    = (float*)alloc((size_t)B_ * 512 * 4);
    __bf16* Tb   = (__bf16*)alloc((size_t)B_ * 512 * 2);

    auto pack = [&](const float* W, int ldw, int col0, int Kpad, int Ksrc, int N,
                    unsigned* dst) {
        int total = (Kpad / 32) * (N / 16) * 256;
        hipLaunchKernelGGL(pack_b_kernel, dim3((total + 255) / 256), dim3(256), 0,
                           stream, W, ldw, col0, Kpad, Ksrc, N, dst);
    };

    // ---- weight packing ----
    pack(comp_w1, 1024, 0, V_, V_, 1024, bp_w1);
    pack(comp_w2, CD_, 0, 1024, 1024, CD_, bp_w2);
    pack(fus_w, HID_, 0, FPAD_, FEAT_, HID_, bp_fus);
    for (int l = 0; l < NL_; l++) {
        const float* in_w  = (const float*)d_in[15 + 12 * l + 0];
        const float* out_w = (const float*)d_in[15 + 12 * l + 2];
        const float* ff_w1 = (const float*)d_in[15 + 12 * l + 6];
        const float* ff_w2 = (const float*)d_in[15 + 12 * l + 8];
        pack(in_w, 3 * HID_, 2 * HID_, HID_, HID_, HID_, bp_wv[l]);  // v slice
        pack(out_w, HID_, 0, HID_, HID_, HID_, bp_wo[l]);
        pack(ff_w1, 512, 0, HID_, HID_, 512, bp_f1[l]);
        pack(ff_w2, HID_, 0, 512, 512, HID_, bp_f2[l]);
    }

    // ---- entropy / sampling / pos features -> combined[:,256:320] ----
    hipLaunchKernelGGL(features_kernel, dim3(B_), dim3(256), 0, stream,
                       logits, positions, pos_emb, comb);

    // ---- G1: logits @ comp_w1 (f32 A, on-the-fly bf16) ----
    hipLaunchKernelGGL((wmma_gemm_kernel<true>), dim3(1024 / 64, B_ / 128), dim3(256),
                       0, stream, (const void*)logits, V_, bp_w1, H1,
                       (const float*)nullptr, B_, 1024, V_);
    hipLaunchKernelGGL(ln_gelu_kernel, dim3(B_), dim3(256), 0, stream,
                       H1, 1024, comp_b1, comp_ln1_g, comp_ln1_b, 1024,
                       h1b, 1024, 0, (float*)nullptr, 0);

    // ---- G2: h1 @ comp_w2 -> combined[:,0:256] ----
    hipLaunchKernelGGL((wmma_gemm_kernel<false>), dim3(CD_ / 64, B_ / 128), dim3(256),
                       0, stream, (const void*)h1b, 1024, bp_w2, T,
                       (const float*)nullptr, B_, CD_, 1024);
    hipLaunchKernelGGL(ln_gelu_kernel, dim3(B_), dim3(256), 0, stream,
                       T, CD_, comp_b2, comp_ln2_g, comp_ln2_b, CD_,
                       comb, FPAD_, 0, (float*)nullptr, 0);

    // ---- G3: combined @ fus_w -> X ----
    hipLaunchKernelGGL((wmma_gemm_kernel<false>), dim3(HID_ / 64, B_ / 128), dim3(256),
                       0, stream, (const void*)comb, FPAD_, bp_fus, T,
                       (const float*)nullptr, B_, HID_, FPAD_);
    hipLaunchKernelGGL(ln_gelu_kernel, dim3(B_), dim3(256), 0, stream,
                       T, HID_, fus_b, fus_ln_g, fus_ln_b, HID_,
                       Xb, HID_, 0, X, HID_);

    // ---- transformer layers (seqlen==1 => attention output == v) ----
    for (int l = 0; l < NL_; l++) {
        const float* in_b  = (const float*)d_in[15 + 12 * l + 1];
        const float* out_b = (const float*)d_in[15 + 12 * l + 3];
        const float* ln1_g = (const float*)d_in[15 + 12 * l + 4];
        const float* ln1_b = (const float*)d_in[15 + 12 * l + 5];
        const float* ff_b1 = (const float*)d_in[15 + 12 * l + 7];
        const float* ff_b2 = (const float*)d_in[15 + 12 * l + 9];
        const float* ln2_g = (const float*)d_in[15 + 12 * l + 10];
        const float* ln2_b = (const float*)d_in[15 + 12 * l + 11];

        hipLaunchKernelGGL((wmma_gemm_kernel<false>), dim3(HID_ / 64, B_ / 128),
                           dim3(256), 0, stream, (const void*)Xb, HID_, bp_wv[l], T,
                           in_b + 2 * HID_, B_, HID_, HID_);
        hipLaunchKernelGGL(cast_act_kernel, dim3((B_ * HID_ + 255) / 256), dim3(256),
                           0, stream, T, B_ * HID_, 0, Tb);
        hipLaunchKernelGGL((wmma_gemm_kernel<false>), dim3(HID_ / 64, B_ / 128),
                           dim3(256), 0, stream, (const void*)Tb, HID_, bp_wo[l], T,
                           (const float*)nullptr, B_, HID_, HID_);
        hipLaunchKernelGGL(resid_ln_kernel, dim3(B_), dim3(HID_), 0, stream,
                           X, T, out_b, ln1_g, ln1_b, Xb);
        hipLaunchKernelGGL((wmma_gemm_kernel<false>), dim3(512 / 64, B_ / 128),
                           dim3(256), 0, stream, (const void*)Xb, HID_, bp_f1[l], T,
                           ff_b1, B_, 512, HID_);
        hipLaunchKernelGGL(cast_act_kernel, dim3((B_ * 512 + 255) / 256), dim3(256),
                           0, stream, T, B_ * 512, 1, Tb);
        hipLaunchKernelGGL((wmma_gemm_kernel<false>), dim3(HID_ / 64, B_ / 128),
                           dim3(256), 0, stream, (const void*)Tb, 512, bp_f2[l], T,
                           (const float*)nullptr, B_, HID_, 512);
        hipLaunchKernelGGL(resid_ln_kernel, dim3(B_), dim3(HID_), 0, stream,
                           X, T, ff_b2, ln2_g, ln2_b, Xb);
    }

    // ---- output heads ----
    hipLaunchKernelGGL(heads_kernel, dim3(B_ / 256), dim3(256), 0, stream,
                       X, hall_w1, hall_b1, hall_w2, hall_b2,
                       scale_w1, scale_b1, scale_w2, scale_b2, out);
}
